// _VQVAE_25469156065330
// MI455X (gfx1250) — matrix operand order repaired and checked by reading, not compiled, not run on hardware
//
#include <hip/hip_runtime.h>
#include <stdint.h>

#define C_CH   192
#define T_LEN  8192
#define B_SZ   16
#define KTOT   576       // C_CH * 3 taps
#define NCODES 128

// LDS slab geometry: 16 row-pairs x 268-dword stride; columns t-n_blk+4 live
// at [4..259], halo t=n_blk-1 at 3 and t=n_blk+256 at 260. 8*268 % 64 == 32
// -> lane halves of a B-fragment read disjoint bank windows.
#define XSTRIDE 268
#define XBASE   4

typedef __attribute__((ext_vector_type(16))) __bf16 v16bf;
typedef __attribute__((ext_vector_type(8)))  float  v8f;

union FragBF { v16bf v; uint32_t u[8]; };

__device__ __forceinline__ uint16_t f2b(float f) {
  uint32_t u = __float_as_uint(f);
  uint32_t r = u + 0x7FFFu + ((u >> 16) & 1u);   // round-to-nearest-even
  return (uint16_t)(r >> 16);
}
__device__ __forceinline__ float b2f(uint16_t h) {
  return __uint_as_float(((uint32_t)h) << 16);
}

// ---------------------------------------------------------------- converts --
__global__ __launch_bounds__(256) void k_cvt_x(const float* __restrict__ x,
                                               uint16_t* __restrict__ o, int n8) {
  int i = blockIdx.x * 256 + threadIdx.x;        // 8 elements per thread
  if (i < n8) {
    const float4* xp = (const float4*)x + (size_t)i * 2;
    float4 a = xp[0], b = xp[1];
    uint4 P;
    P.x = f2b(a.x) | ((uint32_t)f2b(a.y) << 16);
    P.y = f2b(a.z) | ((uint32_t)f2b(a.w) << 16);
    P.z = f2b(b.x) | ((uint32_t)f2b(b.y) << 16);
    P.w = f2b(b.z) | ((uint32_t)f2b(b.w) << 16);
    ((uint4*)o)[i] = P;
  }
}

struct W5 { const float* p[5]; };

// Repack 5 conv weights (co,ci,dk) fp32 -> (co, k=dk*192+ci) bf16; also zero
// the loss / histogram accumulators for this launch.
__global__ __launch_bounds__(256) void k_cvt_w(W5 w, uint16_t* __restrict__ wb,
                                               float* __restrict__ loss_acc,
                                               unsigned* __restrict__ counts) {
  if (blockIdx.x == 0) {
    if (threadIdx.x < NCODES) counts[threadIdx.x] = 0u;
    if (threadIdx.x == NCODES) *loss_acc = 0.0f;
  }
  const int per = C_CH * KTOT;                 // 110592 per layer
  int i = blockIdx.x * 256 + threadIdx.x;
  if (i < 5 * per) {
    int L = i / per, r = i % per;
    int co = r / KTOT, k = r % KTOT;
    int dk = k / C_CH, ci = k % C_CH;
    wb[i] = f2b(w.p[L][(co * C_CH + ci) * 3 + dk]);
  }
}

// ------------------------------------------------------------- conv as GEMM --
// Y[co,t] = sum_k W[co,k] * X[ci(k), t+dk(k)-1],  k = dk*192 + ci.
// Workgroup: 64(M) x 256(N) tile, 8 waves each doing 32x64 via 2x4 WMMA accums.
// K loop = 6 channel-block stages; each stage stages one 32x258 X slab in LDS
// (b128 loads, packed (k_even|k_odd<<16) dwords) reused by all 3 taps.
__global__ __launch_bounds__(256, 1) void k_conv_wmma(
    const uint16_t* __restrict__ xin,   // bf16 (B,C,T)
    const uint16_t* __restrict__ wk,    // bf16 (192, 576)
    const float*    __restrict__ bias,  // (192)
    uint16_t*       __restrict__ out_bf,  // bf16 (B,C,T) or null
    float*          __restrict__ out_f32, // f32  (B,C,T) or null
    int do_relu)
{
  __shared__ uint32_t ldsX[16 * XSTRIDE];   // ~16.8 KB

  const int tid  = threadIdx.x;
  const int lane = tid & 31;
  const int wv   = tid >> 5;
  const int wm   = wv >> 2;          // 0..1  (M half of the tile)
  const int wn   = wv & 3;           // 0..3  (N quarter)
  const int b    = blockIdx.z;
  const int m_blk = blockIdx.x * 64;
  const int n_blk = blockIdx.y * 256;
  const int nlo  = lane & 15;
  const int hh   = lane >> 4;

  const v8f vzero = {0.f, 0.f, 0.f, 0.f, 0.f, 0.f, 0.f, 0.f};
  v8f acc[2][4];
#pragma unroll
  for (int mi = 0; mi < 2; ++mi)
#pragma unroll
    for (int ni = 0; ni < 4; ++ni) acc[mi][ni] = vzero;

  const uint16_t* xb = xin + (size_t)b * C_CH * T_LEN;
  const int p  = tid >> 4;   // row pair 0..15 for the LDS stage
  const int cg = tid & 15;   // column group (16 cols each)

  for (int cb = 0; cb < 6; ++cb) {
    const int cibase = cb * 32;
    const uint16_t* r0 = xb + (size_t)(cibase + 2 * p) * T_LEN;
    const uint16_t* r1 = r0 + T_LEN;

    __syncthreads();                 // prior stage's LDS reads complete
    // ---- bulk: 256 aligned columns, b128 in / packed b128 out ----
#pragma unroll
    for (int h = 0; h < 2; ++h) {
      const int c0 = cg * 16 + h * 8;
      uint4 a  = *(const uint4*)(r0 + n_blk + c0);
      uint4 bq = *(const uint4*)(r1 + n_blk + c0);
      uint4 P0, P1;
      P0.x = (a.x & 0xFFFFu) | (bq.x << 16);
      P0.y = (a.x >> 16)     | (bq.x & 0xFFFF0000u);
      P0.z = (a.y & 0xFFFFu) | (bq.y << 16);
      P0.w = (a.y >> 16)     | (bq.y & 0xFFFF0000u);
      P1.x = (a.z & 0xFFFFu) | (bq.z << 16);
      P1.y = (a.z >> 16)     | (bq.z & 0xFFFF0000u);
      P1.z = (a.w & 0xFFFFu) | (bq.w << 16);
      P1.w = (a.w >> 16)     | (bq.w & 0xFFFF0000u);
      *(uint4*)&ldsX[p * XSTRIDE + XBASE + c0]     = P0;
      *(uint4*)&ldsX[p * XSTRIDE + XBASE + c0 + 4] = P1;
    }
    // ---- halo columns t = n_blk-1 and t = n_blk+256 ('same' padding) ----
    if (tid < 32) {
      const int pe   = tid >> 1;
      const int side = tid & 1;
      const int t    = n_blk + (side ? 256 : -1);
      const uint16_t* e0 = xb + (size_t)(cibase + 2 * pe) * T_LEN;
      uint32_t lo = 0, hi = 0;
      if ((unsigned)t < (unsigned)T_LEN) { lo = e0[t]; hi = e0[t + T_LEN]; }
      ldsX[pe * XSTRIDE + (side ? XBASE + 256 : XBASE - 1)] = lo | (hi << 16);
    }
    // prefetch next channel block's rows into cache while we compute
    if (cb < 5) __builtin_prefetch(r0 + 32 * T_LEN + n_blk + cg * 16, 0, 0);
    __syncthreads();

#pragma unroll
    for (int dk = 0; dk < 3; ++dk) {
      // A fragments (16x32 bf16): lane m = lane&15; lane-half picks K
      // 0-15 vs 16-31; 2 consecutive K per VGPR.
      FragBF a[2];
#pragma unroll
      for (int mi = 0; mi < 2; ++mi) {
        const uint16_t* wrow = wk +
            (size_t)(m_blk + wm * 32 + mi * 16 + nlo) * KTOT + dk * C_CH + cibase;
#pragma unroll
        for (int v = 0; v < 8; ++v) {
          int kp = 2 * v + (hh ? 16 : 0);
          a[mi].u[v] = *(const uint32_t*)(wrow + kp);  // dword-aligned pair
        }
      }
      // B fragments (32x16 bf16): N = lane&15, lane-half picks K 0-15/16-31,
      // VGPR v holds K = khalf+2v, +1 -> one packed LDS dword.
#pragma unroll
      for (int ni = 0; ni < 4; ++ni) {
        FragBF bf;
        const int nb = wn * 64 + ni * 16 + nlo + dk + XBASE - 1;
        const int kr = hh ? 8 : 0;
#pragma unroll
        for (int v = 0; v < 8; ++v) bf.u[v] = ldsX[(kr + v) * XSTRIDE + nb];
#pragma unroll
        for (int mi = 0; mi < 2; ++mi)
          acc[mi][ni] = __builtin_amdgcn_wmma_f32_16x16x32_bf16(
              false, a[mi].v, false, bf.v, (short)0, acc[mi][ni], false, false);
      }
    }
  }

  // Epilogue: C/D layout -> VGPR r is row M=r (lanes 0-15) / r+8 (lanes 16-31)
#pragma unroll
  for (int mi = 0; mi < 2; ++mi) {
#pragma unroll
    for (int r = 0; r < 8; ++r) {
      const int M = m_blk + wm * 32 + mi * 16 + r + (hh ? 8 : 0);
      const float bv = bias[M];
#pragma unroll
      for (int ni = 0; ni < 4; ++ni) {
        const int col = n_blk + wn * 64 + ni * 16 + nlo;
        float v = acc[mi][ni][r] + bv;
        if (do_relu) v = fmaxf(v, 0.0f);
        const size_t o = ((size_t)b * C_CH + M) * T_LEN + col;
        if (out_bf)  out_bf[o]  = f2b(v);
        if (out_f32) out_f32[o] = v;
      }
    }
  }
}

// ----------------------------------------------------------------------- VQ --
// One token per thread; 128-code argmin over d = ||z||^2 + ||e||^2 - 2 z.e,
// emb tiled through LDS in two 64-code blocks (keeps static LDS < 64 KB).
__global__ __launch_bounds__(256) void k_vq(const uint16_t* __restrict__ ze,
                                            const float* __restrict__ emb,
                                            float* __restrict__ loss_acc,
                                            unsigned* __restrict__ counts) {
  __shared__ __align__(16) float embs[64 * C_CH];   // 48 KB
  __shared__ float ee[64];
  __shared__ unsigned cnt[NCODES];
  __shared__ float red[256];

  const int tid = threadIdx.x;
  if (tid < NCODES) cnt[tid] = 0u;

  const int token = blockIdx.x * 256 + tid;       // 512 blocks * 256 = 131072
  const int b = token >> 13;                      // T = 8192
  const int t = token & (T_LEN - 1);
  const uint16_t* zp = ze + (size_t)b * C_CH * T_LEN + t;

  float z[C_CH];
  float zz = 0.0f;
#pragma unroll
  for (int c = 0; c < C_CH; ++c) {
    z[c] = b2f(zp[(size_t)c * T_LEN]);            // coalesced over t
    zz = fmaf(z[c], z[c], zz);
  }

  float dmin = 3.4e38f;
  int imin = 0;
  for (int kb = 0; kb < 2; ++kb) {
    __syncthreads();
    for (int j = tid; j < 64 * C_CH; j += 256) embs[j] = emb[kb * 64 * C_CH + j];
    __syncthreads();
    if (tid < 64) {
      float s = 0.0f;
      for (int c = 0; c < C_CH; ++c) { float e = embs[tid * C_CH + c]; s = fmaf(e, e, s); }
      ee[tid] = s;
    }
    __syncthreads();
    for (int k = 0; k < 64; ++k) {
      float dot = 0.0f;
#pragma unroll
      for (int c = 0; c < C_CH; c += 4) {
        float4 e = *(const float4*)&embs[k * C_CH + c];   // LDS broadcast
        dot = fmaf(z[c],     e.x, dot);
        dot = fmaf(z[c + 1], e.y, dot);
        dot = fmaf(z[c + 2], e.z, dot);
        dot = fmaf(z[c + 3], e.w, dot);
      }
      float d = zz + ee[k] - 2.0f * dot;
      if (d < dmin) { dmin = d; imin = kb * 64 + k; }     // first-index ties
    }
  }
  __syncthreads();
  atomicAdd(&cnt[imin], 1u);
  red[tid] = dmin;                                        // = ||z - e_min||^2
  __syncthreads();
  for (int s = 128; s > 0; s >>= 1) {
    if (tid < s) red[tid] += red[tid + s];
    __syncthreads();
  }
  if (tid == 0) atomicAdd(loss_acc, red[0]);
  if (tid < NCODES) atomicAdd(&counts[tid], cnt[tid]);
}

__global__ void k_fin(const float* __restrict__ loss_acc,
                      const unsigned* __restrict__ counts,
                      float* __restrict__ out, int out_size) {
  __shared__ float red[NCODES];
  const int t = threadIdx.x;
  const float n = (float)(B_SZ * T_LEN);
  float pr = (float)counts[t] / n;
  red[t] = pr * logf(pr + 1e-10f);
  __syncthreads();
  for (int s = 64; s > 0; s >>= 1) {
    if (t < s) red[t] += red[t + s];
    __syncthreads();
  }
  if (t == 0) {
    out[0] = 1.25f * loss_acc[0] / ((float)(B_SZ * T_LEN) * (float)C_CH);
    out[out_size - 1] = expf(-red[0]);
  }
}

// -------------------------------------------------------------------- launch --
extern "C" void kernel_launch(void* const* d_in, const int* in_sizes, int n_in,
                              void* d_out, int out_size, void* d_ws, size_t ws_size,
                              hipStream_t stream) {
  (void)in_sizes; (void)n_in; (void)ws_size;
  const float* x      = (const float*)d_in[0];
  const float* enc_w1 = (const float*)d_in[1];
  const float* enc_b1 = (const float*)d_in[2];
  const float* enc_w2 = (const float*)d_in[3];
  const float* enc_b2 = (const float*)d_in[4];
  const float* emb    = (const float*)d_in[5];
  const float* dec_w1 = (const float*)d_in[6];
  const float* dec_b1 = (const float*)d_in[7];
  const float* dec_w2 = (const float*)d_in[8];
  const float* dec_b2 = (const float*)d_in[9];
  const float* dec_w3 = (const float*)d_in[10];
  const float* dec_b3 = (const float*)d_in[11];

  char* ws = (char*)d_ws;
  const size_t ACT = (size_t)B_SZ * C_CH * T_LEN;        // 25,165,824 elems
  uint16_t* wb       = (uint16_t*)ws;                    // 1,105,920 B
  float*    loss_acc = (float*)(ws + 1179648u);
  unsigned* counts   = (unsigned*)(ws + 1179648u + 64u);
  uint16_t* actA     = (uint16_t*)(ws + (size_t)(2u << 20));
  uint16_t* actB     = (uint16_t*)(ws + (size_t)(2u << 20) + ACT * 2u);

  W5 w5 = {{enc_w1, enc_w2, dec_w1, dec_w2, dec_w3}};
  k_cvt_w<<<(5 * C_CH * KTOT + 255) / 256, 256, 0, stream>>>(w5, wb, loss_acc, counts);
  k_cvt_x<<<(int)(ACT / 8 / 256), 256, 0, stream>>>(x, actA, (int)(ACT / 8));

  const dim3 cg(C_CH / 64, T_LEN / 256, B_SZ);           // (3, 32, 16)
  const int LW = C_CH * KTOT;                            // 110592
  // encoder
  k_conv_wmma<<<cg, 256, 0, stream>>>(actA, wb + 0 * LW, enc_b1, actB, nullptr, 1);
  k_conv_wmma<<<cg, 256, 0, stream>>>(actB, wb + 1 * LW, enc_b2, actA, nullptr, 0); // z_e
  // vector quantize (scalars only; decoder consumes z_e per the reference)
  k_vq<<<512, 256, 0, stream>>>(actA, emb, loss_acc, counts);
  // decoder
  k_conv_wmma<<<cg, 256, 0, stream>>>(actA, wb + 2 * LW, dec_b1, actB, nullptr, 1);
  k_conv_wmma<<<cg, 256, 0, stream>>>(actB, wb + 3 * LW, dec_b2, actA, nullptr, 1);
  k_conv_wmma<<<cg, 256, 0, stream>>>(actA, wb + 4 * LW, dec_b3, nullptr,
                                      ((float*)d_out) + 1, 0);
  k_fin<<<1, NCODES, 0, stream>>>(loss_acc, counts, (float*)d_out, out_size);
}
